// BinaryConv_42125039239539
// MI455X (gfx1250) — 8-bit matrix operand order repaired and checked by reading, compiled, not run on hardware
//
#include <hip/hip_runtime.h>
#include <stdint.h>

typedef __attribute__((ext_vector_type(16))) int   v16i;
typedef __attribute__((ext_vector_type(8)))  float v8f;
typedef int av4i __attribute__((vector_size(16)));

#define NIMG 32
#define CIN 256
#define COUT 256
#define HWD 56
#define SPATIAL (HWD*HWD)
#define MTOT (NIMG*SPATIAL)

#define Y_OFF   ((size_t)0)
#define BX_OFF  ((size_t)MTOT*COUT*4)
#define BW_OFF  (BX_OFF + (size_t)MTOT*CIN)
#define SUM_OFF (BW_OFF + (size_t)9*CIN*COUT)
#define SC_OFF  (SUM_OFF + (size_t)2048)

#if defined(__HIP_DEVICE_COMPILE__) && defined(__has_builtin)
#if __has_builtin(__builtin_amdgcn_global_load_async_to_lds_b128) && \
    __has_builtin(__builtin_amdgcn_s_wait_asynccnt)
#define HAVE_ASYNC_LDS 1
#endif
#endif

__global__ void k_binw(const float* __restrict__ w, unsigned char* __restrict__ bw8) {
  int idx = blockIdx.x * 256 + threadIdx.x;
  if (idx >= 9 * CIN * 64) return;
  int tap = idx / (CIN * 64);
  int rem = idx - tap * (CIN * 64);
  int co = rem >> 6;
  int ci4 = (rem & 63) * 4;
  unsigned int packed = 0;
#pragma unroll
  for (int k = 0; k < 4; ++k) {
    float f = w[((size_t)co * CIN + ci4 + k) * 9 + tap];
    unsigned int b = (f > 0.f) ? 0x38u : (f < 0.f ? 0xB8u : 0u);
    packed |= b << (8 * k);
  }
  *(unsigned int*)(bw8 + (size_t)tap * CIN * COUT + (size_t)co * CIN + ci4) = packed;
}

__global__ void k_binx(const float* __restrict__ x, unsigned char* __restrict__ bx8) {
  __shared__ unsigned char tile[64 * 68];
  int b = blockIdx.x;
  int ct = b & 3;
  int st = (b >> 2) % 49;
  int img = b / 196;
  int cbase = ct * 64, sbase = st * 64;
  int t = threadIdx.x;
  {
    int i = t >> 2, j0 = (t & 3) * 16;
    const float4* src = (const float4*)(x + ((size_t)(img * CIN + cbase + i)) * SPATIAL + sbase + j0);
#pragma unroll
    for (int q = 0; q < 4; ++q) {
      float4 v = src[q];
      float fv[4] = { v.x, v.y, v.z, v.w };
      unsigned int p = 0;
#pragma unroll
      for (int k = 0; k < 4; ++k) {
        unsigned int bb = (fv[k] > 0.f) ? 0x38u : (fv[k] < 0.f ? 0xB8u : 0u);
        p |= bb << (8 * k);
      }
      *(unsigned int*)(&tile[i * 68 + j0 + q * 4]) = p;
    }
  }
  __syncthreads();
  {
    int j = t >> 2, i0 = (t & 3) * 16;
    unsigned int d[4];
#pragma unroll
    for (int q = 0; q < 4; ++q) {
      unsigned int p = 0;
#pragma unroll
      for (int k = 0; k < 4; ++k)
        p |= (unsigned int)tile[(i0 + q * 4 + k) * 68 + j] << (8 * k);
      d[q] = p;
    }
    uint4 outv = make_uint4(d[0], d[1], d[2], d[3]);
    *(uint4*)(bx8 + ((size_t)(img * SPATIAL + sbase + j)) * CIN + cbase + i0) = outv;
  }
}

#define XT_STRIDE 144
__global__ __launch_bounds__(256, 1)
void k_conv(const unsigned char* __restrict__ bx8,
            const unsigned char* __restrict__ bw8,
            float* __restrict__ y,
            float* __restrict__ bsum) {
  __shared__ unsigned char xt[232 * XT_STRIDE];
  const int b    = blockIdx.x;
  const int img  = b / 28;
  const int h0   = (b % 28) * 2;
  const int tid  = threadIdx.x;
  const int wv   = tid >> 5;
  const int lane = tid & 31;
  const int half = lane >> 4;
  const int half8 = half * 8;
  const int nb   = wv * 32;

  v8f acc[7][2];
#pragma unroll
  for (int fm = 0; fm < 7; ++fm)
#pragma unroll
    for (int fn = 0; fn < 2; ++fn)
#pragma unroll
      for (int e = 0; e < 8; ++e) acc[fm][fn][e] = 0.f;

  int pos0[7];
#pragma unroll
  for (int fm = 0; fm < 7; ++fm) {
    int m = fm * 16 + (lane & 15);
    int r = m / 56, c = m - r * 56;
    pos0[fm] = (r * 58 + c) * XT_STRIDE;
  }

  for (int chunk = 0; chunk < 2; ++chunk) {
    const int cinb = chunk << 7;
    for (int idx = tid; idx < 232 * 8; idx += 256) {
      int pos = idx >> 3, seg = idx & 7;
      int row = pos / 58, col = pos - row * 58;
      int hh = h0 - 1 + row, ww = col - 1;
      unsigned char* ldst = &xt[pos * XT_STRIDE + seg * 16];
      if ((unsigned)hh < 56u && (unsigned)ww < 56u) {
        const unsigned char* gsrc =
            bx8 + ((size_t)(img * SPATIAL + hh * 56 + ww)) * CIN + cinb + seg * 16;
#ifdef HAVE_ASYNC_LDS
        __builtin_amdgcn_global_load_async_to_lds_b128((av4i*)gsrc, (av4i*)ldst, 0, 0);
#else
        *(uint4*)ldst = *(const uint4*)gsrc;
#endif
      } else {
        *(uint4*)ldst = make_uint4(0u, 0u, 0u, 0u);
      }
    }
#ifdef HAVE_ASYNC_LDS
    __builtin_amdgcn_s_wait_asynccnt(0);
#endif
    __syncthreads();

    for (int tap = 0; tap < 9; ++tap) {
      const int kh = tap / 3, kw = tap - kh * 3;
      const unsigned char* bp = bw8 + (size_t)tap * CIN * COUT + (size_t)(nb + (lane & 15)) * CIN + cinb + half8;
      v16i b0, b1;
#pragma unroll
      for (int g = 0; g < 2; ++g)
#pragma unroll
        for (int jj = 0; jj < 4; ++jj) {
          int2 r0 = *(const int2*)(bp + g * 64 + jj * 16);
          int2 r1 = *(const int2*)(bp + (size_t)16 * CIN + g * 64 + jj * 16);
          b0[g * 8 + jj * 2] = r0.x; b0[g * 8 + jj * 2 + 1] = r0.y;
          b1[g * 8 + jj * 2] = r1.x; b1[g * 8 + jj * 2 + 1] = r1.y;
        }
      const int tofs = (kh * 58 + kw) * XT_STRIDE + half8;

      v16i a_cur, a_nxt;
      {
        const unsigned char* ap = &xt[pos0[0] + tofs];
#pragma unroll
        for (int g = 0; g < 2; ++g)
#pragma unroll
          for (int jj = 0; jj < 4; ++jj) {
            int2 q = *(const int2*)(ap + g * 64 + jj * 16);
            a_cur[g * 8 + jj * 2]     = q.x;
            a_cur[g * 8 + jj * 2 + 1] = q.y;
          }
      }
#pragma unroll
      for (int fm = 0; fm < 7; ++fm) {
        if (fm < 6) {
          const unsigned char* ap = &xt[pos0[fm + 1] + tofs];
#pragma unroll
          for (int g = 0; g < 2; ++g)
#pragma unroll
            for (int jj = 0; jj < 4; ++jj) {
              int2 q = *(const int2*)(ap + g * 64 + jj * 16);
              a_nxt[g * 8 + jj * 2]     = q.x;
              a_nxt[g * 8 + jj * 2 + 1] = q.y;
            }
        }
        acc[fm][0] = __builtin_amdgcn_wmma_f32_16x16x128_fp8_fp8(a_cur, b0, (short)0, acc[fm][0], false, false);
        acc[fm][1] = __builtin_amdgcn_wmma_f32_16x16x128_fp8_fp8(a_cur, b1, (short)0, acc[fm][1], false, false);
        if (fm < 6) a_cur = a_nxt;
      }
    }
    __syncthreads();
  }

  const int base_m = img * SPATIAL + h0 * 56;
  const int ncol = lane & 15;
#pragma unroll
  for (int fn = 0; fn < 2; ++fn) {
    int col = nb + fn * 16 + ncol;
    float s1 = 0.f, s2 = 0.f;
#pragma unroll
    for (int fm = 0; fm < 7; ++fm) {
      int mrow = base_m + fm * 16 + half8;
#pragma unroll
      for (int v = 0; v < 8; ++v) {
        float val = acc[fm][fn][v];
        y[(size_t)(mrow + v) * COUT + col] = val;
        s1 += val;
        s2 += val * val;
      }
    }
    s1 += __shfl_xor(s1, 16, 32);
    s2 += __shfl_xor(s2, 16, 32);
    if (half == 0) {
      atomicAdd(&bsum[col], s1);
      atomicAdd(&bsum[256 + col], s2);
    }
  }
}

__global__ void k_bnstat(const float* __restrict__ bsum, const float* __restrict__ gamma,
                         const float* __restrict__ beta, float* __restrict__ sc) {
  int c = threadIdx.x;
  const float inv_n = 1.0f / (float)MTOT;
  float mean = bsum[c] * inv_n;
  float var  = bsum[256 + c] * inv_n - mean * mean;
  float s = gamma[c] * rsqrtf(var + 1e-5f);
  sc[c]       = s;
  sc[256 + c] = beta[c] - mean * s;
}

__global__ void k_final(const float* __restrict__ y, const float* __restrict__ sc,
                        float* __restrict__ out) {
  __shared__ float tile[64 * 65];
  int b = blockIdx.x;
  int ct = b & 3;
  int st = (b >> 2) % 49;
  int img = b / 196;
  int cbase = ct * 64, sbase = st * 64;
  int t = threadIdx.x;
  {
    int i = t >> 2, j0 = (t & 3) * 16;
    const float4* src = (const float4*)(y + ((size_t)(img * SPATIAL + sbase + i)) * COUT + cbase + j0);
#pragma unroll
    for (int q = 0; q < 4; ++q) {
      float4 v = src[q];
      tile[i * 65 + j0 + q * 4 + 0] = v.x;
      tile[i * 65 + j0 + q * 4 + 1] = v.y;
      tile[i * 65 + j0 + q * 4 + 2] = v.z;
      tile[i * 65 + j0 + q * 4 + 3] = v.w;
    }
  }
  __syncthreads();
  {
    int j = t >> 2, i0 = (t & 3) * 16;
    int c = cbase + j;
    float s  = sc[c];
    float sh = sc[256 + c];
    float* dst = out + ((size_t)(img * COUT + c)) * SPATIAL + sbase + i0;
#pragma unroll
    for (int k = 0; k < 16; ++k)
      dst[k] = tile[(i0 + k) * 65 + j] * s + sh;
  }
}

extern "C" void kernel_launch(void* const* d_in, const int* in_sizes, int n_in,
                              void* d_out, int out_size, void* d_ws, size_t ws_size,
                              hipStream_t stream) {
  (void)in_sizes; (void)n_in; (void)out_size; (void)ws_size;
  const float* x     = (const float*)d_in[0];
  const float* w     = (const float*)d_in[1];
  const float* gamma = (const float*)d_in[2];
  const float* beta  = (const float*)d_in[3];
  float* out = (float*)d_out;

  unsigned char* ws  = (unsigned char*)d_ws;
  float*         y    = (float*)(ws + Y_OFF);
  unsigned char* bx8  = ws + BX_OFF;
  unsigned char* bw8  = ws + BW_OFF;
  float*         bsum = (float*)(ws + SUM_OFF);
  float*         sc   = (float*)(ws + SC_OFF);

  (void)hipMemsetAsync(bsum, 0, 512 * sizeof(float), stream);
  k_binw<<<576, 256, 0, stream>>>(w, bw8);
  k_binx<<<6272, 256, 0, stream>>>(x, bx8);
  k_conv<<<896, 256, 0, stream>>>(bx8, bw8, y, bsum);
  k_bnstat<<<1, 256, 0, stream>>>(bsum, gamma, beta, sc);
  k_final<<<6272, 256, 0, stream>>>(y, sc, out);
}
